// NTM_13159779795433
// MI455X (gfx1250) — compile-verified, matrix operand order checked by reading, not run on hardware
//
#include <hip/hip_runtime.h>
#include <hip/hip_bf16.h>

// ---------------- constants ----------------
#define Bsz 1024
#define Nn  512
#define Mm  128
#define CW  256
#define RH  4
#define WH  4
#define RLEN 134        // M + 6
#define WLEN 390        // 3M + 6
#define G3  768         // 3*CW
#define EPSf 1e-12f

typedef __attribute__((ext_vector_type(16))) __bf16 v16bf;
typedef __attribute__((ext_vector_type(8)))  float  v8f;

// ---------------- helpers ----------------
__device__ __forceinline__ float sigmoidf_(float x) { return 1.f / (1.f + expf(-x)); }
__device__ __forceinline__ float softplusf_(float x) { return (x > 20.f) ? x : log1pf(expf(x)); }

// Pack 16 floats (4 x float4, k = base+0..7 then base+16..23) into a CDNA5 16-bit WMMA fragment,
// scaled by s (s = 0 zeroes an out-of-range W tile without exec-mask branches).
__device__ __forceinline__ v16bf pack16(float4 q0, float4 q1, float4 q2, float4 q3, float s) {
    v16bf v;
    v[0]  = (__bf16)(q0.x * s); v[1]  = (__bf16)(q0.y * s);
    v[2]  = (__bf16)(q0.z * s); v[3]  = (__bf16)(q0.w * s);
    v[4]  = (__bf16)(q1.x * s); v[5]  = (__bf16)(q1.y * s);
    v[6]  = (__bf16)(q1.z * s); v[7]  = (__bf16)(q1.w * s);
    v[8]  = (__bf16)(q2.x * s); v[9]  = (__bf16)(q2.y * s);
    v[10] = (__bf16)(q2.z * s); v[11] = (__bf16)(q2.w * s);
    v[12] = (__bf16)(q3.x * s); v[13] = (__bf16)(q3.y * s);
    v[14] = (__bf16)(q3.z * s); v[15] = (__bf16)(q3.w * s);
    return v;
}

// ---------------- WMMA GEMM: C[1024 x N] = A[1024 x K] @ W[N x K]^T (+bias) (+=) ---
// 128 threads = 4 waves; each wave computes one 16x16 tile via v_wmma_f32_16x16x32_bf16.
// CDNA5 16-bit fragment: lane L holds row (L&15); kShift = (L&16)?8:0;
// element e -> k = kShift + e (e<8), k = 16 + kShift + (e-8) (e>=8)  => contiguous b128 loads.
template<int K>
__global__ void __launch_bounds__(128) wmma_gemm_bias(
    const float* __restrict__ A, const float* __restrict__ W,
    const float* __restrict__ bias, float* __restrict__ C,
    int N, int accum)
{
    const int lane    = threadIdx.x & 31;
    const int wave    = threadIdx.x >> 5;
    const int rowBase = blockIdx.y * 16;
    const int colBase = (blockIdx.x * 4 + wave) * 16;

    const int ar     = rowBase + (lane & 15);
    const int wrRaw  = colBase + (lane & 15);
    const int wr     = (wrRaw < N) ? wrRaw : 0;       // clamp, zero via scale
    const float wsc  = (wrRaw < N) ? 1.f : 0.f;
    const int kShift = (lane & 16) ? 8 : 0;

    const float* Arow = A + (size_t)ar * K + kShift;
    const float* Wrow = W + (size_t)wr * K + kShift;

    v8f c;
#pragma unroll
    for (int r = 0; r < 8; ++r) c[r] = 0.f;

#pragma unroll
    for (int k0 = 0; k0 < K; k0 += 32) {
        const float4* a4 = (const float4*)(Arow + k0);
        const float4* w4 = (const float4*)(Wrow + k0);
        // k = base+0..7 and base+16..23 (base = k0 + kShift)
        float4 aq0 = a4[0], aq1 = a4[1], aq2 = a4[4], aq3 = a4[5];
        float4 wq0 = w4[0], wq1 = w4[1], wq2 = w4[4], wq3 = w4[5];
        v16bf a = pack16(aq0, aq1, aq2, aq3, 1.f);
        v16bf b = pack16(wq0, wq1, wq2, wq3, wsc);
        c = __builtin_amdgcn_wmma_f32_16x16x32_bf16(false, a, false, b,
                                                    (short)0, c, false, false);
    }

    // C layout: vgpr r -> m = rowBase + r + ((lane&16)?8:0), n = colBase + (lane&15)
    const int m0 = rowBase + ((lane & 16) ? 8 : 0);
    const int n  = colBase + (lane & 15);
    if (n < N) {
        float bv = bias ? bias[n] : 0.f;
#pragma unroll
        for (int r = 0; r < 8; ++r) {
            size_t idx = (size_t)(m0 + r) * N + n;
            float v = c[r] + bv;
            if (accum) v += C[idx];
            C[idx] = v;
        }
    }
}

// ---------------- GRU gate fusion: o = (1-z)*n + z*h ----------------
__global__ void gru_combine(const float* __restrict__ gi, const float* __restrict__ gh,
                            const float* __restrict__ hidden, float* __restrict__ o)
{
    int idx = blockIdx.x * blockDim.x + threadIdx.x;
    if (idx >= Bsz * CW) return;
    int b = idx >> 8, cc = idx & (CW - 1);
    const float* gib = gi + (size_t)b * G3;
    const float* ghb = gh + (size_t)b * G3;
    float r = sigmoidf_(gib[cc] + ghb[cc]);
    float z = sigmoidf_(gib[cc + CW] + ghb[cc + CW]);
    float nn = tanhf(gib[cc + 2 * CW] + r * ghb[cc + 2 * CW]);
    float h = hidden[idx];
    o[idx] = (1.f - z) * nn + z * h;
}

// ---------------- per-batch mega kernel: memory[b] resident in 320KB LDS --------
// LDS floats: mem 65536 | kn 1024 | cosw 4096 | red 512 | par 64 | er 512 | ad 512
#define L_MEM 0
#define L_KN  65536
#define L_COS (65536 + 1024)
#define L_RED (65536 + 1024 + 4096)
#define L_PAR (65536 + 1024 + 4096 + 512)
#define L_ER  (65536 + 1024 + 4096 + 512 + 64)
#define L_AD  (65536 + 1024 + 4096 + 512 + 64 + 512)
#define L_TOT (65536 + 1024 + 4096 + 512 + 64 + 512 + 512)

__global__ void __launch_bounds__(512, 1) ntm_mega(
    const float* __restrict__ memory, const float* __restrict__ rp,
    const float* __restrict__ wp, const float* __restrict__ read_w,
    const float* __restrict__ write_w, float* __restrict__ rv,
    float* __restrict__ new_mem)
{
    extern __shared__ float smem[];
    const int b    = blockIdx.x;
    const int tid  = threadIdx.x;
    const int lane = tid & 31;
    const int wave = tid >> 5;        // 16 waves of 32 (wave32)

    float*  memL  = smem + L_MEM;
    float*  knL   = smem + L_KN;
    float*  cosw  = smem + L_COS;
    float*  red   = smem + L_RED;
    float*  par   = smem + L_PAR;
    float*  erL   = smem + L_ER;
    float*  adL   = smem + L_AD;
    float4* memL4 = (float4*)memL;

    // ---- stage 1: memory[b] (256 KB) -> LDS via gfx1250 async copy if available ---
    const float* gmem = memory + (size_t)b * (Nn * Mm);
#if __has_builtin(__builtin_amdgcn_global_load_async_to_lds_b128)
    {
        typedef int v4i_ __attribute__((vector_size(16)));
        typedef __attribute__((address_space(1))) v4i_ v4i_g;   // global
        typedef __attribute__((address_space(3))) v4i_ v4i_l;   // LDS
        v4i_g* gsrc = (v4i_g*)gmem;          // C-style cast: addrspacecast + de-const
        v4i_l* ldst = (v4i_l*)memL;
        for (int i = tid; i < Nn * Mm / 4; i += 512)
            __builtin_amdgcn_global_load_async_to_lds_b128(gsrc + i, ldst + i, 0, 0);
#if __has_builtin(__builtin_amdgcn_s_wait_asynccnt)
        __builtin_amdgcn_s_wait_asynccnt(0);
#else
        asm volatile("s_wait_asynccnt 0" ::: "memory");
#endif
    }
#else
    {
        const float4* gmem4 = (const float4*)gmem;
        for (int i = tid; i < Nn * Mm / 4; i += 512) memL4[i] = gmem4[i];
    }
#endif

    for (int i = tid; i < 8 * Mm; i += 512) {       // 8 raw keys (4 read + 4 write)
        int h = i >> 7, m = i & (Mm - 1);
        knL[i] = (h < 4) ? rp[(size_t)b * (RH * RLEN) + h * RLEN + m]
                         : wp[(size_t)b * (WH * WLEN) + (h - 4) * WLEN + m];
    }
    if (tid < 8) {                                  // per-head scalar params
        int h = tid;
        const float* p = (h < 4) ? (rp + (size_t)b * (RH * RLEN) + h * RLEN + Mm)
                                 : (wp + (size_t)b * (WH * WLEN) + (h - 4) * WLEN + Mm);
        float beta = softplusf_(p[0]);
        float g    = sigmoidf_(p[1]);
        float s0 = p[2], s1 = p[3], s2 = p[4];
        float mx = fmaxf(s0, fmaxf(s1, s2));
        float e0 = expf(s0 - mx), e1 = expf(s1 - mx), e2 = expf(s2 - mx);
        float es = e0 + e1 + e2;
        par[h * 8 + 0] = beta;     par[h * 8 + 1] = g;
        par[h * 8 + 2] = e0 / es;  par[h * 8 + 3] = e1 / es;  par[h * 8 + 4] = e2 / es;
        par[h * 8 + 5] = 1.f + softplusf_(p[5]);
    }
    {                                               // erase (sigmoid) / add (tanh)
        int h = tid >> 7, m = tid & (Mm - 1);       // 512 threads = 4 heads x 128
        const float* wph = wp + (size_t)b * (WH * WLEN) + h * WLEN;
        erL[tid] = sigmoidf_(wph[Mm + 6 + m]);
        adL[tid] = tanhf(wph[2 * Mm + 6 + m]);
    }
    __syncthreads();

    // ---- stage 2: key norms (wave-per-key), then normalize keys ----
    if (wave < 8) {
        const float* k = knL + wave * Mm;
        float s = 0.f;
        for (int j = lane; j < Mm; j += 32) s += k[j] * k[j];
#pragma unroll
        for (int o = 16; o; o >>= 1) s += __shfl_xor(s, o, 32);
        if (lane == 0) red[wave] = 1.f / (sqrtf(s) + EPSf);
    }
    __syncthreads();
    for (int i = tid; i < 8 * Mm; i += 512) knL[i] *= red[i >> 7];
    __syncthreads();

    // ---- stage 3: row inv-norm + 8 cosine dots per row (wave-per-row) ----
    const float4* kn4 = (const float4*)knL;
    for (int row = wave; row < Nn; row += 16) {
        float4 v = memL4[row * (Mm / 4) + lane];            // conflict-free b128 LDS
        float s = v.x * v.x + v.y * v.y + v.z * v.z + v.w * v.w;
#pragma unroll
        for (int o = 16; o; o >>= 1) s += __shfl_xor(s, o, 32);
        float inv = 1.f / (sqrtf(s) + EPSf);
#pragma unroll
        for (int h = 0; h < 8; ++h) {
            float4 kk = kn4[h * (Mm / 4) + lane];
            float d = v.x * kk.x + v.y * kk.y + v.z * kk.z + v.w * kk.w;
#pragma unroll
            for (int o = 16; o; o >>= 1) d += __shfl_xor(d, o, 32);
            if (lane == 0) cosw[h * Nn + row] = d * inv;
        }
    }
    __syncthreads();

    // ---- stage 4: addressing per head (thread tid owns slot n=tid) ----
    for (int h = 0; h < 8; ++h) {
        float beta = par[h * 8 + 0], g = par[h * 8 + 1];
        float s0 = par[h * 8 + 2], s1 = par[h * 8 + 3], s2 = par[h * 8 + 4];
        float gam = par[h * 8 + 5];
        int n = tid;
        float e = beta * cosw[h * Nn + n];
        red[tid] = e; __syncthreads();
        for (int s = 256; s > 0; s >>= 1) { if (tid < s) red[tid] = fmaxf(red[tid], red[tid + s]); __syncthreads(); }
        float mx = red[0]; __syncthreads();
        float ex = expf(e - mx);
        red[tid] = ex; __syncthreads();
        for (int s = 256; s > 0; s >>= 1) { if (tid < s) red[tid] += red[tid + s]; __syncthreads(); }
        float sum = red[0]; __syncthreads();
        float wt = ex / sum;
        float prev = (h < 4) ? read_w[((size_t)b * RH + h) * Nn + n]
                             : write_w[((size_t)b * WH + (h - 4)) * Nn + n];
        wt = g * wt + (1.f - g) * prev;
        cosw[h * Nn + n] = wt; __syncthreads();
        float sh = s0 * cosw[h * Nn + ((n + 1) & (Nn - 1))]
                 + s1 * wt
                 + s2 * cosw[h * Nn + ((n + Nn - 1) & (Nn - 1))];
        float pw = expf(gam * logf(sh + EPSf));             // (shifted+eps)^gamma
        red[tid] = pw; __syncthreads();
        for (int s = 256; s > 0; s >>= 1) { if (tid < s) red[tid] += red[tid + s]; __syncthreads(); }
        float T = red[0]; __syncthreads();
        cosw[h * Nn + n] = pw / (T + EPSf);
        __syncthreads();
    }

    // ---- stage 5: read vectors rv[b, h*128+m] = sum_n w_r[h,n] * mem[n,m] ----
    {
        int h = tid >> 7, m = tid & (Mm - 1);               // 512 threads = 4x128
        float acc = 0.f;
        for (int n = 0; n < Nn; ++n) acc += cosw[h * Nn + n] * memL[n * Mm + m];
        rv[(size_t)b * (RH * Mm) + tid] = acc;
    }

    // ---- stage 6: erase/add update (4 write heads, sequential), write new_mem ---
    float4* out4 = (float4*)(new_mem + (size_t)b * (Nn * Mm));
    for (int i = tid; i < Nn * Mm / 4; i += 512) {
        int n = i >> 5;                 // 32 float4 per row
        int m4 = (i & 31) * 4;
        float4 v = memL4[i];
#pragma unroll
        for (int h = 0; h < 4; ++h) {
            float wt = cosw[(4 + h) * Nn + n];
            const float* eh = erL + h * Mm + m4;
            const float* ah = adL + h * Mm + m4;
            v.x = v.x * (1.f - wt * eh[0]) + wt * ah[0];
            v.y = v.y * (1.f - wt * eh[1]) + wt * ah[1];
            v.z = v.z * (1.f - wt * eh[2]) + wt * ah[2];
            v.w = v.w * (1.f - wt * eh[3]) + wt * ah[3];
        }
        out4[i] = v;
    }
}

// ---------------- launch ----------------
extern "C" void kernel_launch(void* const* d_in, const int* in_sizes, int n_in,
                              void* d_out, int out_size, void* d_ws, size_t ws_size,
                              hipStream_t stream)
{
    const float* x       = (const float*)d_in[0];
    const float* memory  = (const float*)d_in[1];
    const float* hidden  = (const float*)d_in[2];
    const float* read_w  = (const float*)d_in[3];
    const float* write_w = (const float*)d_in[4];
    const float* W_ih    = (const float*)d_in[5];
    const float* W_hh    = (const float*)d_in[6];
    const float* b_ih    = (const float*)d_in[7];
    const float* b_hh    = (const float*)d_in[8];
    const float* W_rh    = (const float*)d_in[9];
    const float* b_rh    = (const float*)d_in[10];
    const float* W_wh    = (const float*)d_in[11];
    const float* b_wh    = (const float*)d_in[12];
    const float* W_rm    = (const float*)d_in[13];
    const float* W_out   = (const float*)d_in[14];
    const float* b_out   = (const float*)d_in[15];

    float* ws = (float*)d_ws;
    float* gi = ws;                                   // 1024*768
    float* gh = gi + (size_t)Bsz * G3;                // 1024*768
    float* o  = gh + (size_t)Bsz * G3;                // 1024*256
    float* rp = o  + (size_t)Bsz * CW;                // 1024*536
    float* wp = rp + (size_t)Bsz * (RH * RLEN);       // 1024*1560
    float* rv = wp + (size_t)Bsz * (WH * WLEN);       // 1024*512

    float* y       = (float*)d_out;                   // 1024*128
    float* new_mem = y + (size_t)Bsz * Mm;            // 1024*512*128

    dim3 blk(128);
    // GRU input/hidden projections (N multiple of 64 -> exact grids)
    wmma_gemm_bias<Mm><<<dim3(G3 / 64, Bsz / 16), blk, 0, stream>>>(x,      W_ih, b_ih, gi, G3, 0);
    wmma_gemm_bias<CW><<<dim3(G3 / 64, Bsz / 16), blk, 0, stream>>>(hidden, W_hh, b_hh, gh, G3, 0);
    gru_combine<<<(Bsz * CW) / 256, 256, 0, stream>>>(gi, gh, hidden, o);
    // head parameter projections (ragged N handled by clamp+scale and store guard)
    wmma_gemm_bias<CW><<<dim3((RH * RLEN + 63) / 64, Bsz / 16), blk, 0, stream>>>(o, W_rh, b_rh, rp, RH * RLEN, 0);
    wmma_gemm_bias<CW><<<dim3((WH * WLEN + 63) / 64, Bsz / 16), blk, 0, stream>>>(o, W_wh, b_wh, wp, WH * WLEN, 0);
    // addressing + read vectors + memory update, memory[b] resident in LDS
    size_t smem = (size_t)L_TOT * sizeof(float);      // ~282 KB of the 320 KB WGP LDS
    ntm_mega<<<Bsz, 512, smem, stream>>>(memory, rp, wp, read_w, write_w, rv, new_mem);
    // y = o @ W_out^T + b_out + rv @ W_rm^T
    wmma_gemm_bias<CW><<<dim3(Mm / 64, Bsz / 16), blk, 0, stream>>>(o,  W_out, b_out,   y, Mm, 0);
    wmma_gemm_bias<RH * Mm><<<dim3(Mm / 64, Bsz / 16), blk, 0, stream>>>(rv, W_rm, nullptr, y, Mm, 1);
}
